// AttentionCell_mid_6485400617365
// MI455X (gfx1250) — matrix operand order repaired and checked by reading, not compiled, not run on hardware
//
#include <hip/hip_runtime.h>
#include <hip/hip_bf16.h>

// Shapes
#define Bn 512
#define Tn 256
#define In 512
#define Hn 512
#define CMn 256
#define CBn 256
#define G4H 2048          // 4*H
#define KX 1024           // I + CM + CB
#define KTOT 1536         // KX + H

typedef __attribute__((ext_vector_type(16))) __bf16 v16bf;
typedef __attribute__((ext_vector_type(8)))  __bf16 v8bf;
typedef __attribute__((ext_vector_type(8)))  float  v8f;

// ---- WMMA fragment loaders -------------------------------------------------
// A fragment (bf16 16x32, MxK) from an LDS tile of bf16 rows with stride lda.
// lane L: M = L%16, khalf = L/16; elements j=0..7 -> K = 8*khalf + j,
//                                 j=8..15 -> K = 16 + 8*khalf + (j-8)
__device__ inline v16bf load_a_lds(const __bf16* ldsA, int lda, int row0, int k0, int lane) {
    const int m  = lane & 15;
    const int kh = (lane >> 4) << 3;
    const __bf16* p = ldsA + (row0 + m) * lda + k0 + kh;
    v8bf lo = *(const v8bf*)p;          // K = k0+kh .. +7   (16B aligned)
    v8bf hi = *(const v8bf*)(p + 16);   // K = k0+kh+16 .. +23
    v16bf a;
#pragma unroll
    for (int j = 0; j < 8; ++j) { a[j] = lo[j]; a[j + 8] = hi[j]; }
    return a;
}

// B fragment (bf16 32x16, KxN) where B[k][n] = W[n][k], W row-major [N x ldw] bf16.
// lane L: N = n0 + L%16; j=0..7 -> K = k0 + 8*(L/16) + j, j=8..15 -> K = k0 + 16 + 8*(L/16) + (j-8)
__device__ inline v16bf load_b_global16(const __bf16* __restrict__ W, int ldw,
                                        int n0, int k0, int lane) {
    const int n  = n0 + (lane & 15);
    const int kb = k0 + ((lane >> 4) << 3);
    const __bf16* p = W + (size_t)n * ldw + kb;
    v8bf b0 = *(const v8bf*)(p);        // 16B, K = kb..kb+7
    v8bf b1 = *(const v8bf*)(p + 16);   // 16B, K = kb+16..kb+23
    v16bf b;
#pragma unroll
    for (int j = 0; j < 8; ++j) { b[j] = b0[j]; b[j + 8] = b1[j]; }
    return b;
}

// ---- Kernel 0: fp32 -> bf16 weight mirror (one-time per launch) ------------
__global__ __launch_bounds__(256) void cvt_bf16_kernel(const float* __restrict__ src,
                                                       __bf16* __restrict__ dst, int n) {
    const int stride = gridDim.x * 256;
    for (int i = blockIdx.x * 256 + threadIdx.x; i < n; i += stride)
        dst[i] = (__bf16)src[i];
}

// ---- Kernel 1: h_proj = prev_h @ W_h2h^T + b_h2h  [B,H] --------------------
__global__ __launch_bounds__(256) void hproj_kernel(const float* __restrict__ prev_h,
                                                    const float* __restrict__ W_h2h,
                                                    const float* __restrict__ b_h2h,
                                                    float* __restrict__ hproj) {
    __shared__ float hrow[Hn];
    const int b = blockIdx.x, tid = threadIdx.x;
    hrow[tid]       = prev_h[(size_t)b * Hn + tid];
    hrow[tid + 256] = prev_h[(size_t)b * Hn + tid + 256];
    __syncthreads();
    for (int h = tid; h < Hn; h += 256) {
        const float* w = W_h2h + (size_t)h * Hn;
        float acc = b_h2h[h];
#pragma unroll 4
        for (int k = 0; k < Hn; ++k) acc += hrow[k] * w[k];
        hproj[(size_t)b * Hn + h] = acc;
    }
}

// ---- Kernel 2: fused  e[b,t] = tanh(batch_H@W_i2h^T + h_proj) @ W_score^T --
// Block: 32 rows of (b*T+t), 8 waves = 2 rowtiles x 4 n-columns.
// kt-outer / 8-accumulator blocking: A fragment read once per kt, 8 WMMAs per A.
__global__ __launch_bounds__(256) void attn_score_kernel(const float* __restrict__ batch_H,
                                                         const __bf16* __restrict__ W16,
                                                         const float* __restrict__ hproj,
                                                         const float* __restrict__ W_score,
                                                         float* __restrict__ e_out) {
    __shared__ __bf16 ldsA[32 * 520];         // 32 rows x 512 (+8 pad) bf16
    __shared__ float  eacc[4][32];            // per n-column partial sums
    const int tid = threadIdx.x;
    const int rowbase = blockIdx.x * 32;      // global row = b*T + t
    const int b = rowbase >> 8;               // rows all in same b (32 | 256)

    // Stage A tile: f32 -> bf16, coalesced, read batch_H exactly once.
    for (int idx = tid; idx < 32 * 512; idx += 256) {
        const int r = idx >> 9, c = idx & 511;
        ldsA[r * 520 + c] = (__bf16)batch_H[(size_t)(rowbase + r) * In + c];
    }
    __syncthreads();

    const int wave = tid >> 5, lane = tid & 31;
    const int rowtile = wave >> 2;            // 0..1 (16 rows each)
    const int ncol    = wave & 3;             // n-tile = ncol + 4*i, i = 0..7

    v8f acc[8];
#pragma unroll
    for (int i = 0; i < 8; ++i) acc[i] = (v8f){};

    for (int kt = 0; kt < In / 32; ++kt) {
        const v16bf a = load_a_lds(ldsA, 520, rowtile * 16, kt * 32, lane);
#pragma unroll
        for (int i = 0; i < 8; ++i) {
            const int nt = ncol + 4 * i;
            v16bf bm = load_b_global16(W16, In, nt * 16, kt * 32, lane);
            acc[i] = __builtin_amdgcn_wmma_f32_16x16x32_bf16(false, a, false, bm,
                                                             (short)0, acc[i], false, false);
        }
    }

    float rsum[8];
#pragma unroll
    for (int v = 0; v < 8; ++v) rsum[v] = 0.f;
#pragma unroll
    for (int i = 0; i < 8; ++i) {
        const int n  = (ncol + 4 * i) * 16 + (lane & 15);
        const float hp = hproj[(size_t)b * Hn + n];
        const float ws = W_score[n];
#pragma unroll
        for (int v = 0; v < 8; ++v) rsum[v] += tanhf(acc[i][v] + hp) * ws;
    }
    // Reduce over the 16 lanes of each half-wave (columns n).
#pragma unroll
    for (int v = 0; v < 8; ++v) {
        float s = rsum[v];
        s += __shfl_xor(s, 1);
        s += __shfl_xor(s, 2);
        s += __shfl_xor(s, 4);
        s += __shfl_xor(s, 8);
        rsum[v] = s;
    }
    if ((lane & 15) == 0) {
        const int mbase = rowtile * 16 + ((lane >> 4) << 3);
#pragma unroll
        for (int v = 0; v < 8; ++v) eacc[ncol][mbase + v] = rsum[v];
    }
    __syncthreads();
    if (tid < 32)
        e_out[rowbase + tid] = eacc[0][tid] + eacc[1][tid] + eacc[2][tid] + eacc[3][tid];
}

// ---- Kernel 3: softmax over T per batch -> alpha (third output) ------------
__global__ __launch_bounds__(256) void softmax_kernel(const float* __restrict__ e,
                                                      float* __restrict__ alpha) {
    __shared__ float red[Tn];
    const int b = blockIdx.x, t = threadIdx.x;
    const float x = e[(size_t)b * Tn + t];
    red[t] = x; __syncthreads();
    for (int s = 128; s > 0; s >>= 1) { if (t < s) red[t] = fmaxf(red[t], red[t + s]); __syncthreads(); }
    const float mx = red[0]; __syncthreads();
    const float ex = expf(x - mx);
    red[t] = ex; __syncthreads();
    for (int s = 128; s > 0; s >>= 1) { if (t < s) red[t] += red[t + s]; __syncthreads(); }
    alpha[(size_t)b * Tn + t] = ex / red[0];
}

// ---- Kernel 4: context + pack x16=[context|mid|bot] bf16, h16=bf16(prev_h) -
__global__ __launch_bounds__(256) void context_kernel(const float* __restrict__ batch_H,
                                                      const float* __restrict__ alpha,
                                                      const float* __restrict__ onehots_mid,
                                                      const float* __restrict__ onehots_bot,
                                                      const float* __restrict__ prev_h,
                                                      __bf16* __restrict__ x16,
                                                      __bf16* __restrict__ h16) {
    __shared__ float a_s[Tn];
    const int b = blockIdx.x, tid = threadIdx.x;
    a_s[tid] = alpha[(size_t)b * Tn + tid];
    __syncthreads();
    for (int i = tid; i < In; i += 256) {
        float acc = 0.f;
        const float* src = batch_H + (size_t)b * Tn * In + i;
#pragma unroll 4
        for (int t = 0; t < Tn; ++t) acc += a_s[t] * src[(size_t)t * In];
        x16[(size_t)b * KX + i]  = (__bf16)acc;
        h16[(size_t)b * Hn + i]  = (__bf16)prev_h[(size_t)b * Hn + i];
    }
    x16[(size_t)b * KX + In + tid]        = (__bf16)onehots_mid[(size_t)b * CMn + tid];
    x16[(size_t)b * KX + In + CMn + tid]  = (__bf16)onehots_bot[(size_t)b * CBn + tid];
}

// ---- Kernel 5: gates = [x|h] @ [W_ih|W_hh]^T + b_ih + b_hh  (WMMA) ---------
// kt-outer / 8-accumulator blocking, two N-tile groups per wave.
__global__ __launch_bounds__(256) void gates_kernel(const __bf16* __restrict__ x16,
                                                    const __bf16* __restrict__ h16,
                                                    const __bf16* __restrict__ Wih16,
                                                    const __bf16* __restrict__ Whh16,
                                                    const float* __restrict__ b_ih,
                                                    const float* __restrict__ b_hh,
                                                    float* __restrict__ gates) {
    __shared__ __bf16 ldsA[16 * 1544];        // 16 rows x 1536 (+8 pad) bf16
    const int tid = threadIdx.x;
    const int rowbase = blockIdx.x * 16;
    for (int idx = tid; idx < 16 * KTOT; idx += 256) {
        const int r = idx / KTOT, c = idx % KTOT;
        const __bf16 v = (c < KX) ? x16[(size_t)(rowbase + r) * KX + c]
                                  : h16[(size_t)(rowbase + r) * Hn + (c - KX)];
        ldsA[r * 1544 + c] = v;
    }
    __syncthreads();

    const int wave = tid >> 5, lane = tid & 31;
    for (int g = 0; g < 2; ++g) {
        v8f acc[8];
#pragma unroll
        for (int i = 0; i < 8; ++i) acc[i] = (v8f){};

        for (int kt = 0; kt < KTOT / 32; ++kt) {
            const int k0 = kt * 32;
            const v16bf a = load_a_lds(ldsA, 1544, 0, k0, lane);
#pragma unroll
            for (int i = 0; i < 8; ++i) {
                const int nt = wave + (g * 8 + i) * 8;   // 0..127, disjoint per wave
                v16bf bm = (k0 < KX) ? load_b_global16(Wih16, KX, nt * 16, k0, lane)
                                     : load_b_global16(Whh16, Hn, nt * 16, k0 - KX, lane);
                acc[i] = __builtin_amdgcn_wmma_f32_16x16x32_bf16(false, a, false, bm,
                                                                 (short)0, acc[i], false, false);
            }
        }
#pragma unroll
        for (int i = 0; i < 8; ++i) {
            const int n = (wave + (g * 8 + i) * 8) * 16 + (lane & 15);
            const float bias = b_ih[n] + b_hh[n];
            const int mb = (lane >> 4) << 3;
#pragma unroll
            for (int v = 0; v < 8; ++v)
                gates[(size_t)(rowbase + mb + v) * G4H + n] = acc[i][v] + bias;
        }
    }
}

// ---- Kernel 6: LSTM elementwise -> h_new, c_new ----------------------------
__global__ __launch_bounds__(256) void lstm_kernel(const float* __restrict__ gates,
                                                   const float* __restrict__ prev_c,
                                                   float* __restrict__ h_new,
                                                   float* __restrict__ c_new) {
    const int b = blockIdx.x, tid = threadIdx.x;
    const float* g = gates + (size_t)b * G4H;
    for (int h = tid; h < Hn; h += 256) {
        const float ig = g[h], fg = g[Hn + h], gg = g[2 * Hn + h], og = g[3 * Hn + h];
        const float si = 1.f / (1.f + expf(-ig));
        const float sf = 1.f / (1.f + expf(-fg));
        const float so = 1.f / (1.f + expf(-og));
        const float cn = sf * prev_c[(size_t)b * Hn + h] + si * tanhf(gg);
        c_new[(size_t)b * Hn + h] = cn;
        h_new[(size_t)b * Hn + h] = so * tanhf(cn);
    }
}

extern "C" void kernel_launch(void* const* d_in, const int* in_sizes, int n_in,
                              void* d_out, int out_size, void* d_ws, size_t ws_size,
                              hipStream_t stream) {
    const float* prev_h      = (const float*)d_in[0];
    const float* prev_c      = (const float*)d_in[1];
    const float* batch_H     = (const float*)d_in[2];
    const float* onehots_mid = (const float*)d_in[3];
    const float* onehots_bot = (const float*)d_in[4];
    const float* W_i2h       = (const float*)d_in[5];
    const float* W_h2h       = (const float*)d_in[6];
    const float* b_h2h       = (const float*)d_in[7];
    const float* W_score     = (const float*)d_in[8];
    const float* W_ih        = (const float*)d_in[9];
    const float* W_hh        = (const float*)d_in[10];
    const float* b_ih        = (const float*)d_in[11];
    const float* b_hh        = (const float*)d_in[12];

    // Outputs: h_new [B,H] | c_new [B,H] | alpha [B,T,1]
    float* h_new = (float*)d_out;
    float* c_new = h_new + (size_t)Bn * Hn;
    float* alpha = c_new + (size_t)Bn * Hn;

    // Workspace layout (~14 MB)
    float*  e      = (float*)d_ws;                        // B*T
    float*  hproj  = e + (size_t)Bn * Tn;                 // B*H
    float*  gates  = hproj + (size_t)Bn * Hn;             // B*4H
    __bf16* x16    = (__bf16*)(gates + (size_t)Bn * G4H); // B*KX
    __bf16* h16    = x16 + (size_t)Bn * KX;               // B*H
    __bf16* wi2h16 = h16 + (size_t)Bn * Hn;               // H*I
    __bf16* wih16  = wi2h16 + (size_t)Hn * In;            // 4H*KX
    __bf16* whh16  = wih16 + (size_t)G4H * KX;            // 4H*H

    // One-time bf16 weight mirrors (streamed thousands of times from L2 after).
    cvt_bf16_kernel<<<256, 256, 0, stream>>>(W_i2h, wi2h16, Hn * In);
    cvt_bf16_kernel<<<512, 256, 0, stream>>>(W_ih, wih16, G4H * KX);
    cvt_bf16_kernel<<<512, 256, 0, stream>>>(W_hh, whh16, G4H * Hn);

    hproj_kernel<<<Bn, 256, 0, stream>>>(prev_h, W_h2h, b_h2h, hproj);
    attn_score_kernel<<<(Bn * Tn) / 32, 256, 0, stream>>>(batch_H, wi2h16, hproj, W_score, e);
    softmax_kernel<<<Bn, 256, 0, stream>>>(e, alpha);
    context_kernel<<<Bn, 256, 0, stream>>>(batch_H, alpha, onehots_mid, onehots_bot,
                                           prev_h, x16, h16);
    gates_kernel<<<Bn / 16, 256, 0, stream>>>(x16, h16, wih16, whh16, b_ih, b_hh, gates);
    lstm_kernel<<<Bn, 256, 0, stream>>>(gates, prev_c, h_new, c_new);
}
